// RetrosynthesisDecoder_29351806501168
// MI455X (gfx1250) — matrix operand order repaired and checked by reading, not compiled, not run on hardware
//
#include <hip/hip_runtime.h>

typedef __attribute__((ext_vector_type(16))) _Float16 v16h;
typedef __attribute__((ext_vector_type(8)))  _Float16 v8h;
typedef __attribute__((ext_vector_type(8)))  float    v8f;

#define B_     32
#define T_     128
#define U_     256
#define V_     5000
#define VPAD64 5056   // padded to 64-column multiple for 2x2 tile blocking
#define MROWS  4096   // B*T

__device__ __forceinline__ v16h pack16(v8h lo, v8h hi) {
    v16h r;
#pragma unroll
    for (int i = 0; i < 8; ++i) { r[i] = lo[i]; r[i + 8] = hi[i]; }
    return r;
}

__device__ __forceinline__ v8f wmma_f16(v16h a, v16h b, v8f c) {
    return __builtin_amdgcn_wmma_f32_16x16x32_f16(false, a, false, b, (short)0, c, false, false);
}

__device__ __forceinline__ float sigmoidf_(float x) { return 1.f / (1.f + __expf(-x)); }

// ---------- conversion kernels ----------

// dst[Npad,K] f16 = transpose(src[K,N]) with zero padding rows n>=N
__global__ void transpose_cvt(const float* __restrict__ src, _Float16* __restrict__ dst,
                              int K, int N, int Npad) {
    int idx = blockIdx.x * blockDim.x + threadIdx.x;
    int total = Npad * K;
    if (idx >= total) return;
    int n = idx / K, k = idx % K;
    dst[idx] = (n < N) ? (_Float16)src[(size_t)k * N + n] : (_Float16)0.f;
}

__global__ void cvt_f16(const float* __restrict__ src, _Float16* __restrict__ dst, int n) {
    int idx = blockIdx.x * blockDim.x + threadIdx.x;
    if (idx < n) dst[idx] = (_Float16)src[idx];
}

__global__ void pad_copy(const float* __restrict__ src, float* __restrict__ dst, int n, int npad) {
    int idx = blockIdx.x * blockDim.x + threadIdx.x;
    if (idx < npad) dst[idx] = (idx < n) ? src[idx] : 0.f;
}

// X[t*32+b, u] = f16(emb[ids[b,t], u])   (time-major activations)
__global__ void embed_kernel(const int* __restrict__ ids, const float* __restrict__ emb,
                             _Float16* __restrict__ X) {
    int idx = blockIdx.x * blockDim.x + threadIdx.x;  // MROWS*U_
    if (idx >= MROWS * U_) return;
    int row = idx >> 8, u = idx & 255;
    int t = row >> 5, b = row & 31;
    int id = ids[b * T_ + t];
    X[idx] = (_Float16)emb[(size_t)id * U_ + u];
}

// ---------- tile store helper ----------
// mode 0: C[m*ldc+col]
// mode 1: out row remap (m=t*32+b -> b*T_+t), col guard < ncl
// mode 2: store transposed enc_p: m=b*128+e -> C[(b*256+col)*128+e]
__device__ __forceinline__ void store_tile(float* __restrict__ C, int ldc,
                                           const float* __restrict__ bias,
                                           int mode, int ncl, int mt, int nt,
                                           v8f acc, int hi, int l15) {
    int col = nt * 16 + l15;
    float bv = bias ? bias[col] : 0.f;
#pragma unroll
    for (int r = 0; r < 8; ++r) {
        int row = mt * 16 + hi * 8 + r;
        float v = acc[r] + bv;
        if (mode == 0) {
            C[(size_t)row * ldc + col] = v;
        } else if (mode == 1) {
            if (col < ncl) {
                int b = row & 31, d = row >> 5;
                C[((size_t)b * T_ + d) * ldc + col] = v;
            }
        } else {
            int b = row >> 7, e = row & 127;
            C[((size_t)b * 256 + col) * 128 + e] = v;
        }
    }
}

// ---------- generic f16 WMMA GEMM with 2x2 register blocking ----------
// C = A[M,K] * Bt[N,K]^T + bias. Block = 4 waves covering 64x64; each wave 32x32.
// M, N must be multiples of 32 per wave region (all call sites: 64-multiples).
__global__ void __launch_bounds__(128)
gemm_f16(const _Float16* __restrict__ A, int lda,
         const _Float16* __restrict__ Bt, int ldb,
         float* __restrict__ C, int ldc,
         const float* __restrict__ bias,
         int Mtiles, int Ntiles, int K, int mode, int ncl) {
    int wave = threadIdx.x >> 5, lane = threadIdx.x & 31;
    int mt0 = blockIdx.y * 4 + (wave >> 1) * 2;
    int nt0 = blockIdx.x * 4 + (wave & 1) * 2;
    if (mt0 >= Mtiles || nt0 >= Ntiles) return;
    int hi = lane >> 4, l15 = lane & 15;

    const _Float16* A0 = A + (size_t)(mt0 * 16 + l15) * lda + hi * 8;
    const _Float16* A1 = A0 + (size_t)16 * lda;
    const _Float16* B0 = Bt + (size_t)(nt0 * 16 + l15) * ldb + hi * 8;
    const _Float16* B1 = B0 + (size_t)16 * ldb;

    v8f acc00 = {0.f, 0.f, 0.f, 0.f, 0.f, 0.f, 0.f, 0.f};
    v8f acc01 = acc00, acc10 = acc00, acc11 = acc00;
    for (int kc = 0; kc < K; kc += 32) {
        v16h af0 = pack16(*(const v8h*)(A0 + kc), *(const v8h*)(A0 + kc + 16));
        v16h af1 = pack16(*(const v8h*)(A1 + kc), *(const v8h*)(A1 + kc + 16));
        v16h bf0 = pack16(*(const v8h*)(B0 + kc), *(const v8h*)(B0 + kc + 16));
        v16h bf1 = pack16(*(const v8h*)(B1 + kc), *(const v8h*)(B1 + kc + 16));
        acc00 = wmma_f16(af0, bf0, acc00);
        acc01 = wmma_f16(af0, bf1, acc01);
        acc10 = wmma_f16(af1, bf0, acc10);
        acc11 = wmma_f16(af1, bf1, acc11);
    }
    store_tile(C, ldc, bias, mode, ncl, mt0,     nt0,     acc00, hi, l15);
    store_tile(C, ldc, bias, mode, ncl, mt0,     nt0 + 1, acc01, hi, l15);
    store_tile(C, ldc, bias, mode, ncl, mt0 + 1, nt0,     acc10, hi, l15);
    store_tile(C, ldc, bias, mode, ncl, mt0 + 1, nt0 + 1, acc11, hi, l15);
}

// ---------- persistent single-WGP LSTM recurrence ----------
// Z: precomputed x@W+b, time-major [T][32][1024]. Ut: [1024,256] f16 (transposed).
// h kept in LDS f16; c in registers; gates transposed through 128KB LDS.
__global__ void __launch_bounds__(1024, 1)
lstm_scan(const float* __restrict__ Z, const _Float16* __restrict__ Ut,
          const int* __restrict__ ids, const float* __restrict__ h0,
          const float* __restrict__ c0, int useInit,
          _Float16* __restrict__ Xout) {
    __shared__ _Float16 hbuf[32 * 256];      // 16 KB
    __shared__ float    zbuf[32 * 1024];     // 128 KB
    int tid = threadIdx.x;
    int wave = tid >> 5, lane = tid & 31, hi = lane >> 4, l15 = lane & 15;

    float creg[8];
#pragma unroll
    for (int j = 0; j < 8; ++j) {
        int cell = j * 1024 + tid;
        creg[j] = useInit ? c0[cell] : 0.f;
        hbuf[cell] = (_Float16)(useInit ? h0[cell] : 0.f);
    }
    __syncthreads();

    for (int t = 0; t < T_; ++t) {
        // Prefetch next timestep's Z slab (128KB = 1024 x 128B lines, one per thread)
        if (t + 1 < T_)
            __builtin_prefetch(Z + (size_t)(t + 1) * 32 * 1024 + tid * 32, 0, 1);

        // Phase A: R = h @ U  (32x1024, K=256) -> zbuf. 128 tiles / 32 waves.
#pragma unroll
        for (int ti = 0; ti < 4; ++ti) {
            int mt = ti & 1;
            int ct = wave * 2 + (ti >> 1);
            const _Float16* Ab = hbuf + (mt * 16 + l15) * 256 + hi * 8;
            const _Float16* Bb = Ut + (size_t)(ct * 16 + l15) * 256 + hi * 8;
            v8f acc = {0.f, 0.f, 0.f, 0.f, 0.f, 0.f, 0.f, 0.f};
#pragma unroll
            for (int kc = 0; kc < 256; kc += 32) {
                v8h a0 = *(const v8h*)(Ab + kc);
                v8h a1 = *(const v8h*)(Ab + kc + 16);
                v8h b0 = *(const v8h*)(Bb + kc);
                v8h b1 = *(const v8h*)(Bb + kc + 16);
                acc = wmma_f16(pack16(a0, a1), pack16(b0, b1), acc);
            }
            int col = ct * 16 + l15;
            int rb = mt * 16 + hi * 8;
#pragma unroll
            for (int r = 0; r < 8; ++r) zbuf[(rb + r) * 1024 + col] = acc[r];
        }
        __syncthreads();

        // Phase B: gates + cell update. 8192 cells / 1024 threads.
        const float* Zt = Z + (size_t)t * 32 * 1024;
#pragma unroll
        for (int j = 0; j < 8; ++j) {
            int cell = j * 1024 + tid;
            int b = cell >> 8, u = cell & 255;
            const float* zg = Zt + b * 1024;
            const float* zl = zbuf + b * 1024;
            float iv = sigmoidf_(zg[u] + zl[u]);
            float fv = sigmoidf_(zg[256 + u] + zl[256 + u]);
            float gv = tanhf(zg[512 + u] + zl[512 + u]);
            float ov = sigmoidf_(zg[768 + u] + zl[768 + u]);
            float cn = fv * creg[j] + iv * gv;
            float hn = ov * tanhf(cn);
            bool m = ids[b * T_ + t] != 0;
            float hp = (float)hbuf[cell];
            float ho = m ? hn : hp;
            creg[j] = m ? cn : creg[j];
            hbuf[cell] = (_Float16)ho;
            Xout[((size_t)t * 32 + b) * U_ + u] = (_Float16)ho;
        }
        __syncthreads();
    }
}

// ---------- fused attention: score + softmax + context + concat ----------
// block = one (b,d); 128 threads (= encoder positions)
__global__ void __launch_bounds__(128)
attn_kernel(const float* __restrict__ dec_p,   // [t*32+b, 256]
            const float* __restrict__ enc_pT,  // [b, u, e]
            const float* __restrict__ enc,     // [b, e, u] fp32 input
            const float* __restrict__ attnV,   // [256]
            const _Float16* __restrict__ Xfin, // [t*32+b, 256]
            _Float16* __restrict__ C16) {      // [t*32+b, 512]
    __shared__ float decs[256], vv[256], wsm[128], red[128];
    int rb = blockIdx.x;
    int b = rb >> 7, d = rb & 127;
    int rowt = d * 32 + b;
    int tid = threadIdx.x;

    decs[tid] = dec_p[(size_t)rowt * 256 + tid];
    decs[tid + 128] = dec_p[(size_t)rowt * 256 + tid + 128];
    vv[tid] = attnV[tid];
    vv[tid + 128] = attnV[tid + 128];
    __syncthreads();

    // score for encoder position e = tid
    float s = 0.f;
    const float* ep = enc_pT + (size_t)b * 256 * 128 + tid;
#pragma unroll 4
    for (int u = 0; u < 256; ++u)
        s += tanhf(decs[u] + ep[(size_t)u * 128]) * vv[u];

    // softmax over 128 positions
    red[tid] = s; __syncthreads();
    for (int st = 64; st > 0; st >>= 1) {
        if (tid < st) red[tid] = fmaxf(red[tid], red[tid + st]);
        __syncthreads();
    }
    float mx = red[0]; __syncthreads();
    float e = __expf(s - mx);
    red[tid] = e; __syncthreads();
    for (int st = 64; st > 0; st >>= 1) {
        if (tid < st) red[tid] += red[tid + st];
        __syncthreads();
    }
    float inv = 1.f / red[0]; __syncthreads();
    wsm[tid] = e * inv;
    __syncthreads();

    // context: ctx[u] = sum_e w[e] * enc[b,e,u]; thread handles u=tid, tid+128
    float cA = 0.f, cB = 0.f;
    const float* eb = enc + (size_t)b * 128 * 256;
#pragma unroll 4
    for (int ee = 0; ee < 128; ++ee) {
        float we = wsm[ee];
        cA += we * eb[(size_t)ee * 256 + tid];
        cB += we * eb[(size_t)ee * 256 + tid + 128];
    }
    _Float16* crow = C16 + (size_t)rowt * 512;
    crow[tid]       = Xfin[(size_t)rowt * 256 + tid];
    crow[tid + 128] = Xfin[(size_t)rowt * 256 + tid + 128];
    crow[256 + tid] = (_Float16)cA;
    crow[384 + tid] = (_Float16)cB;
}

// ---------- in-place row softmax on d_out ----------
__global__ void __launch_bounds__(256)
softmax_rows(float* __restrict__ data, int ncols) {
    __shared__ float red[256];
    int tid = threadIdx.x;
    float* rowp = data + (size_t)blockIdx.x * ncols;
    float mx = -3.4e38f;
    for (int c = tid; c < ncols; c += 256) mx = fmaxf(mx, rowp[c]);
    red[tid] = mx; __syncthreads();
    for (int st = 128; st > 0; st >>= 1) {
        if (tid < st) red[tid] = fmaxf(red[tid], red[tid + st]);
        __syncthreads();
    }
    mx = red[0]; __syncthreads();
    float s = 0.f;
    for (int c = tid; c < ncols; c += 256) {
        float ev = __expf(rowp[c] - mx);
        rowp[c] = ev; s += ev;
    }
    red[tid] = s; __syncthreads();
    for (int st = 128; st > 0; st >>= 1) {
        if (tid < st) red[tid] += red[tid + st];
        __syncthreads();
    }
    float inv = 1.f / red[0];
    for (int c = tid; c < ncols; c += 256) rowp[c] *= inv;
}

// ---------- host launch ----------
extern "C" void kernel_launch(void* const* d_in, const int* in_sizes, int n_in,
                              void* d_out, int out_size, void* d_ws, size_t ws_size,
                              hipStream_t stream) {
    const int*   ids = (const int*)d_in[0];
    const float* h0  = (const float*)d_in[1];
    const float* c0  = (const float*)d_in[2];
    const float* enc = (const float*)d_in[3];
    const float* emb = (const float*)d_in[4];
    const float* Ws  = (const float*)d_in[5];
    const float* Us  = (const float*)d_in[6];
    const float* bs  = (const float*)d_in[7];
    const float* aW1 = (const float*)d_in[8];
    const float* aW2 = (const float*)d_in[9];
    const float* aV  = (const float*)d_in[10];
    const float* dW  = (const float*)d_in[11];
    const float* db  = (const float*)d_in[12];
    float* out = (float*)d_out;

    char* wsb = (char*)d_ws;
    size_t off = 0;
    auto alloc = [&](size_t bytes) -> void* {
        void* p = wsb + off;
        off += (bytes + 255) & ~(size_t)255;
        return p;
    };
    _Float16* W16   = (_Float16*)alloc((size_t)4 * 1024 * 256 * 2);
    _Float16* U16   = (_Float16*)alloc((size_t)4 * 1024 * 256 * 2);
    _Float16* aW1t  = (_Float16*)alloc((size_t)256 * 256 * 2);
    _Float16* aW2t  = (_Float16*)alloc((size_t)256 * 256 * 2);
    _Float16* dW16  = (_Float16*)alloc((size_t)VPAD64 * 512 * 2);
    float*    dbias = (float*)   alloc((size_t)VPAD64 * 4);
    _Float16* enc16 = (_Float16*)alloc((size_t)MROWS * U_ * 2);
    _Float16* Xa    = (_Float16*)alloc((size_t)MROWS * U_ * 2);
    _Float16* Xb    = (_Float16*)alloc((size_t)MROWS * U_ * 2);
    float*    Zbuf  = (float*)   alloc((size_t)MROWS * 1024 * 4);
    float*    encpT = (float*)   alloc((size_t)MROWS * U_ * 4);
    float*    decp  = (float*)   alloc((size_t)MROWS * U_ * 4);
    _Float16* C16   = (_Float16*)alloc((size_t)MROWS * 512 * 2);

    // weight conversions (transpose to [N,K] f16)
    {
        int n = 1024 * 256;
        for (int l = 0; l < 4; ++l) {
            transpose_cvt<<<(n + 255) / 256, 256, 0, stream>>>(Ws + (size_t)l * 256 * 1024, W16 + (size_t)l * 1024 * 256, 256, 1024, 1024);
            transpose_cvt<<<(n + 255) / 256, 256, 0, stream>>>(Us + (size_t)l * 256 * 1024, U16 + (size_t)l * 1024 * 256, 256, 1024, 1024);
        }
        int n2 = 256 * 256;
        transpose_cvt<<<(n2 + 255) / 256, 256, 0, stream>>>(aW1, aW1t, 256, 256, 256);
        transpose_cvt<<<(n2 + 255) / 256, 256, 0, stream>>>(aW2, aW2t, 256, 256, 256);
        int n3 = VPAD64 * 512;
        transpose_cvt<<<(n3 + 255) / 256, 256, 0, stream>>>(dW, dW16, 512, V_, VPAD64);
        pad_copy<<<(VPAD64 + 255) / 256, 256, 0, stream>>>(db, dbias, V_, VPAD64);
        int n4 = MROWS * U_;
        cvt_f16<<<(n4 + 255) / 256, 256, 0, stream>>>(enc, enc16, n4);
        embed_kernel<<<(n4 + 255) / 256, 256, 0, stream>>>(ids, emb, Xa);
    }

    // 4 LSTM layers: big time-parallel GEMM + persistent recurrence
    _Float16* Xcur = Xa;
    _Float16* Xnxt = Xb;
    for (int l = 0; l < 4; ++l) {
        // Z = X @ W + b : M=4096, N=1024, K=256 -> grid 16 x 64 (64x64 per block)
        gemm_f16<<<dim3(16, 64), 128, 0, stream>>>(
            Xcur, U_, W16 + (size_t)l * 1024 * 256, U_, Zbuf, 1024,
            bs + (size_t)l * 1024, MROWS / 16, 64, U_, 0, 1024);
        lstm_scan<<<1, 1024, 0, stream>>>(
            Zbuf, U16 + (size_t)l * 1024 * 256, ids, h0, c0, (l == 0) ? 1 : 0, Xnxt);
        _Float16* tmp = Xcur; Xcur = Xnxt; Xnxt = tmp;
    }

    // attention projections: M=4096, N=256 -> grid 4 x 64
    gemm_f16<<<dim3(4, 64), 128, 0, stream>>>(enc16, U_, aW1t, U_, encpT, 0, nullptr,
                                              MROWS / 16, 16, U_, 2, 256);
    gemm_f16<<<dim3(4, 64), 128, 0, stream>>>(Xcur, U_, aW2t, U_, decp, U_, nullptr,
                                              MROWS / 16, 16, U_, 0, 256);
    // fused score/softmax/context/concat
    attn_kernel<<<MROWS, 128, 0, stream>>>(decp, encpT, enc, aV, Xcur, C16);

    // logits straight into d_out (remapped to [b,t,v]); N=5056 padded -> grid 79 x 64
    gemm_f16<<<dim3(79, 64), 128, 0, stream>>>(C16, 512, dW16, 512, out, V_, dbias,
                                               MROWS / 16, VPAD64 / 16, 512, 1, V_);
    softmax_rows<<<MROWS, 256, 0, stream>>>(out, V_);
}
